// Qwen3MoeSWAttention_16690242912517
// MI455X (gfx1250) — matrix operand order repaired and checked by reading, not compiled, not run on hardware
//
#include <hip/hip_runtime.h>

#define HIDDEN 2048
#define NH 32
#define NKV 4
#define D 128
#define T_SEQ 2048
#define WINDOW 512
#define QKVN ((NH + 2 * NKV) * D) /* 5120 */
#define EPS 1e-6f
#define THETA 1000000.0f
#define SCALE_QK 0.08838834764831845f /* 128^-0.5 */

typedef __bf16 bf16;
typedef __attribute__((ext_vector_type(16))) __bf16 v16bf;
typedef __attribute__((ext_vector_type(8)))  __bf16 v8bf;
typedef __attribute__((ext_vector_type(4)))  __bf16 v4bf;
typedef __attribute__((ext_vector_type(8)))  float  v8f;

union Frag16 { v16bf v; v8bf h[2]; };
union Cvt16  { float4 f; bf16 b[16]; };

static __device__ __forceinline__ v8f wmma_bf16(v16bf a, v16bf b, v8f c) {
  return __builtin_amdgcn_wmma_f32_16x16x32_bf16(false, a, false, b, (short)0, c,
                                                 false, false);
}

// Async DMA: 16 bytes global -> LDS per lane, tracked by ASYNCcnt (gfx1250).
static __device__ __forceinline__ void async_copy16(unsigned lds_off,
                                                    const void* gaddr) {
  asm volatile("global_load_async_to_lds_b128 %0, %1, off"
               :: "v"(lds_off), "v"((unsigned long long)(uintptr_t)gaddr)
               : "memory");
}
static __device__ __forceinline__ void wait_async0() {
  asm volatile("s_wait_asynccnt 0x0" ::: "memory");
}
static __device__ __forceinline__ unsigned lds_off_of(const void* p) {
  return (unsigned)(uintptr_t)p;  // low 32 bits of generic addr = LDS offset
}

static __device__ __forceinline__ float halfwave_max(float x) {
#pragma unroll
  for (int m = 1; m <= 8; m <<= 1) x = fmaxf(x, __shfl_xor(x, m, 32));
  return x;
}
static __device__ __forceinline__ float halfwave_sum(float x) {
#pragma unroll
  for (int m = 1; m <= 8; m <<= 1) x += __shfl_xor(x, m, 32);
  return x;
}

// ---------------------------------------------------------------------------
// fp32 -> bf16 bulk convert (4 elems/thread)
// ---------------------------------------------------------------------------
__global__ __launch_bounds__(256) void cvt_f32_bf16(const float* __restrict__ x,
                                                    bf16* __restrict__ y, int n4) {
  int i = blockIdx.x * 256 + threadIdx.x;
  if (i >= n4) return;
  float4 f = ((const float4*)x)[i];
  v4bf o;
  o[0] = (bf16)f.x; o[1] = (bf16)f.y; o[2] = (bf16)f.z; o[3] = (bf16)f.w;
  ((v4bf*)y)[i] = o;
}

// ---------------------------------------------------------------------------
// Generic bf16 GEMM: C[M,N] (f32) = A[M,K] * B[K,N], 128x128 block, 8 waves.
// Double-buffered LDS; A tile staged via async-to-LDS DMA, B tile staged
// through registers (transpose scatter) overlapped with WMMA compute.
// ---------------------------------------------------------------------------
__global__ __launch_bounds__(256) void gemm_bf16(const bf16* __restrict__ A,
                                                 const bf16* __restrict__ B,
                                                 float* __restrict__ C,
                                                 int M, int N, int K) {
  __shared__ __align__(16) bf16 As[2][128 * 40];
  __shared__ __align__(16) bf16 Bt[2][128 * 40]; // [n][k] transposed

  const int tid   = threadIdx.x;
  const int wid   = tid >> 5;
  const int lane  = tid & 31;
  const int waveM = wid >> 1;   // 0..3
  const int waveN = wid & 1;    // 0..1
  const int lrow  = lane & 15;
  const int ka    = (lane >> 4) * 8;   // A-frag K base (0 / 8)
  const int kb    = (lane >> 4) * 16;  // unused; B-frag K base is ka (0 / 8)
  const int bM    = blockIdx.y * 128;
  const int bN    = blockIdx.x * 128;

  v8f acc[2][4];
#pragma unroll
  for (int mt = 0; mt < 2; ++mt)
#pragma unroll
    for (int nt = 0; nt < 4; ++nt)
#pragma unroll
      for (int r = 0; r < 8; ++r) acc[mt][nt][r] = 0.f;

  const int rowA = tid >> 1, segA = tid & 1;  // 128 rows x 2 segs of 16
  const int rowB = tid >> 3, segB = tid & 7;  // 32 k-rows x 8 segs of 16
  const bf16* gA = &A[(size_t)(bM + rowA) * K + segA * 16]; // advance by k
  const bf16* gB = &B[(size_t)rowB * N + bN + segB * 16];   // advance by k*N

  // ---- prologue: stage tile 0 ----
  async_copy16(lds_off_of(&As[0][rowA * 40 + segA * 16]), gA);
  {
    Cvt16 t0;
    t0.f = *(const float4*)gB;
#pragma unroll
    for (int e = 0; e < 16; ++e) Bt[0][(segB * 16 + e) * 40 + rowB] = t0.b[e];
  }
  wait_async0();
  __syncthreads();

  int cb = 0;
  for (int k0 = 0; k0 < K; k0 += 32) {
    const int  nb   = cb ^ 1;
    const bool more = (k0 + 32) < K;

    // stage tile t+1: async A copy + B global load into registers
    Cvt16 tN;
    if (more) {
      async_copy16(lds_off_of(&As[nb][rowA * 40 + segA * 16]), gA + (k0 + 32));
      tN.f = *(const float4*)(gB + (size_t)(k0 + 32) * N);
    }

    // compute tile t from buffer cb
    Frag16 af[2], bfr[4];
#pragma unroll
    for (int mt = 0; mt < 2; ++mt) {
      const bf16* p = &As[cb][(waveM * 32 + mt * 16 + lrow) * 40];
      af[mt].h[0] = *(const v8bf*)&p[ka];
      af[mt].h[1] = *(const v8bf*)&p[ka + 16];
    }
#pragma unroll
    for (int nt = 0; nt < 4; ++nt) {
      const bf16* p = &Bt[cb][(waveN * 64 + nt * 16 + lrow) * 40];
      bfr[nt].h[0] = *(const v8bf*)&p[ka];
      bfr[nt].h[1] = *(const v8bf*)&p[ka + 16];
    }
#pragma unroll
    for (int mt = 0; mt < 2; ++mt)
#pragma unroll
      for (int nt = 0; nt < 4; ++nt)
        acc[mt][nt] = wmma_bf16(af[mt].v, bfr[nt].v, acc[mt][nt]);

    // finish staging t+1 (transpose scatter), then sync
    if (more) {
#pragma unroll
      for (int e = 0; e < 16; ++e) Bt[nb][(segB * 16 + e) * 40 + rowB] = tN.b[e];
    }
    wait_async0();
    __syncthreads();
    cb = nb;
  }

  const int rhalf = (lane >> 4) * 8;
#pragma unroll
  for (int mt = 0; mt < 2; ++mt)
#pragma unroll
    for (int nt = 0; nt < 4; ++nt)
#pragma unroll
      for (int r = 0; r < 8; ++r) {
        int row = bM + waveM * 32 + mt * 16 + r + rhalf;
        int col = bN + waveN * 64 + nt * 16 + lrow;
        C[(size_t)row * N + col] = acc[mt][nt][r];
      }
}

// ---------------------------------------------------------------------------
// RMS-norm + RoPE over the fused QKV output; emits bf16 Q/K/V.
// One wave per (token, head-slot). Slot 0..31 = q heads, 32..35 = k, 36..39 = v.
// ---------------------------------------------------------------------------
__global__ __launch_bounds__(256) void norm_rope(const int* __restrict__ positions,
                                                 const float* __restrict__ qkv,
                                                 const float* __restrict__ qw,
                                                 const float* __restrict__ kw,
                                                 bf16* __restrict__ q,
                                                 bf16* __restrict__ k,
                                                 bf16* __restrict__ v) {
  const int t    = blockIdx.x;
  const int wid  = threadIdx.x >> 5;
  const int lane = threadIdx.x & 31;
  const int slot = blockIdx.y * 8 + wid;  // 0..39
  const float* x = qkv + (size_t)t * QKVN + slot * D;
  const int i0   = lane * 2;  // pair index in 0..63

  float2 a = *(const float2*)&x[i0];
  float2 b = *(const float2*)&x[i0 + 64];

  if (slot >= NH + NKV) {  // V: convert only
    bf16* dst = v + ((size_t)t * NKV + (slot - NH - NKV)) * D;
    dst[i0]      = (bf16)a.x;
    dst[i0 + 1]  = (bf16)a.y;
    dst[i0 + 64] = (bf16)b.x;
    dst[i0 + 65] = (bf16)b.y;
    return;
  }

  // per-head RMS norm over D=128
  float ss = a.x * a.x + a.y * a.y + b.x * b.x + b.y * b.y;
#pragma unroll
  for (int m = 16; m >= 1; m >>= 1) ss += __shfl_xor(ss, m, 32);
  const float r = rsqrtf(ss * (1.0f / (float)D) + EPS);

  const float* w = (slot < NH) ? qw : kw;
  float x1a = a.x * r * w[i0];
  float x1b = a.y * r * w[i0 + 1];
  float x2a = b.x * r * w[i0 + 64];
  float x2b = b.y * r * w[i0 + 65];

  const float pos = (float)positions[t];
  const float lt  = __logf(THETA);
  const float f0  = pos * __expf(-((float)i0 * (1.0f / 64.0f)) * lt);
  const float f1  = pos * __expf(-((float)(i0 + 1) * (1.0f / 64.0f)) * lt);
  const float c0 = __cosf(f0), s0 = __sinf(f0);
  const float c1 = __cosf(f1), s1 = __sinf(f1);

  bf16* dst = (slot < NH) ? q + ((size_t)t * NH + slot) * D
                          : k + ((size_t)t * NKV + (slot - NH)) * D;
  dst[i0]      = (bf16)(x1a * c0 - x2a * s0);
  dst[i0 + 1]  = (bf16)(x1b * c1 - x2b * s1);
  dst[i0 + 64] = (bf16)(x2a * c0 + x1a * s0);
  dst[i0 + 65] = (bf16)(x2b * c1 + x1b * s1);
}

// ---------------------------------------------------------------------------
// Sliding-window causal GQA flash attention.
// Grid: (T/128, NH). Block 256 = 8 waves; wave w handles 16 queries.
// Double-buffered 32-key steps; K tile staged via async-to-LDS DMA,
// V tile transposed through registers, both overlapped with WMMA/softmax.
// ---------------------------------------------------------------------------
__global__ __launch_bounds__(256) void attn_kernel(const bf16* __restrict__ q,
                                                   const bf16* __restrict__ kk,
                                                   const bf16* __restrict__ vv,
                                                   bf16* __restrict__ attn) {
  __shared__ __align__(16) bf16 Ks[2][32 * 128];    // [key][d]
  __shared__ __align__(16) bf16 Vt[2][128 * 32];    // [d][key]
  __shared__ __align__(16) bf16 Ps[8 * 16 * 32];    // per-wave P tiles

  const int h    = blockIdx.y;
  const int qb   = blockIdx.x;
  const int tid  = threadIdx.x;
  const int wid  = tid >> 5;
  const int lane = tid & 31;
  const int lrow = lane & 15;
  const int ka   = (lane >> 4) * 8;    // A-frag K base
  const int kb   = (lane >> 4) * 16;   // unused; B-frag K base is ka
  const int rhalf = (lane >> 4) * 8;
  const int q0   = qb * 128 + wid * 16;
  const int kvh  = h >> 3;  // NH/NKV = 8

  // Q fragments (A-layout), resident for whole kernel
  Frag16 qf[4];
  {
    const bf16* qp = q + (size_t)(q0 + lrow) * (NH * D) + (size_t)h * D;
#pragma unroll
    for (int dc = 0; dc < 4; ++dc) {
      qf[dc].h[0] = *(const v8bf*)&qp[dc * 32 + ka];
      qf[dc].h[1] = *(const v8bf*)&qp[dc * 32 + ka + 16];
    }
  }

  float m[8], l[8];
  v8f o[8];
#pragma unroll
  for (int r = 0; r < 8; ++r) { m[r] = -1e30f; l[r] = 0.f; }
#pragma unroll
  for (int nt = 0; nt < 8; ++nt)
#pragma unroll
    for (int r = 0; r < 8; ++r) o[nt][r] = 0.f;

  int jstart = qb * 128 - (WINDOW - 1);
  if (jstart < 0) jstart = 0;
  jstart &= ~31;
  const int jend = qb * 128 + 128;

  const int krow = tid >> 3, kseg = tid & 7;  // 32 keys x 8 d-segs
  const bf16* gK = &kk[(size_t)krow * (NKV * D) + kvh * D + kseg * 16];
  const bf16* gV = &vv[(size_t)krow * (NKV * D) + kvh * D + kseg * 16];

  // ---- prologue: stage tile at jstart ----
  async_copy16(lds_off_of(&Ks[0][krow * 128 + kseg * 16]),
               gK + (size_t)jstart * (NKV * D));
  {
    Cvt16 tv;
    tv.f = *(const float4*)(gV + (size_t)jstart * (NKV * D));
#pragma unroll
    for (int e = 0; e < 16; ++e) Vt[0][(kseg * 16 + e) * 32 + krow] = tv.b[e];
  }
  wait_async0();
  __syncthreads();

  int cb = 0;
  for (int j0 = jstart; j0 < jend; j0 += 32) {
    const int  nb   = cb ^ 1;
    const bool more = (j0 + 32) < jend;

    // stage next tile: async K copy + V global load into registers
    Cvt16 tn;
    if (more) {
      async_copy16(lds_off_of(&Ks[nb][krow * 128 + kseg * 16]),
                   gK + (size_t)(j0 + 32) * (NKV * D));
      tn.f = *(const float4*)(gV + (size_t)(j0 + 32) * (NKV * D));
    }

    // S = Q * K^T for two 16-key column tiles (from buffer cb)
    float st[2][8];
#pragma unroll
    for (int jt = 0; jt < 2; ++jt) {
      v8f s;
#pragma unroll
      for (int r = 0; r < 8; ++r) s[r] = 0.f;
#pragma unroll
      for (int dc = 0; dc < 4; ++dc) {
        Frag16 bfr;
        const bf16* p = &Ks[cb][(jt * 16 + lrow) * 128 + dc * 32];
        bfr.h[0] = *(const v8bf*)&p[ka];
        bfr.h[1] = *(const v8bf*)&p[ka + 16];
        s = wmma_bf16(qf[dc].v, bfr.v, s);
      }
#pragma unroll
      for (int r = 0; r < 8; ++r) {
        int qi = q0 + r + rhalf;
        int kj = j0 + jt * 16 + lrow;
        bool valid = (kj <= qi) && (qi - kj < WINDOW);
        st[jt][r] = valid ? s[r] * SCALE_QK : -1e30f;
      }
    }

    // online softmax update
#pragma unroll
    for (int r = 0; r < 8; ++r) {
      float rmax = halfwave_max(fmaxf(st[0][r], st[1][r]));
      float mn = fmaxf(m[r], rmax);
      float sc = __expf(m[r] - mn);
      m[r] = mn;
      float p0 = __expf(st[0][r] - mn);
      float p1 = __expf(st[1][r] - mn);
      st[0][r] = p0;
      st[1][r] = p1;
      l[r] = l[r] * sc + halfwave_sum(p0 + p1);
#pragma unroll
      for (int nt = 0; nt < 8; ++nt) o[nt][r] *= sc;
    }

    // P (C-layout) -> per-wave LDS -> A-layout fragment
    bf16* pw = &Ps[wid * 16 * 32];
#pragma unroll
    for (int jt = 0; jt < 2; ++jt)
#pragma unroll
      for (int r = 0; r < 8; ++r)
        pw[(r + rhalf) * 32 + jt * 16 + lrow] = (bf16)st[jt][r];
    asm volatile("s_wait_dscnt 0" ::: "memory");
    Frag16 pf;
    pf.h[0] = *(const v8bf*)&pw[lrow * 32 + ka];
    pf.h[1] = *(const v8bf*)&pw[lrow * 32 + ka + 16];

    // O += P * V (from buffer cb)
#pragma unroll
    for (int nt = 0; nt < 8; ++nt) {
      Frag16 bfr;
      const bf16* p = &Vt[cb][(nt * 16 + lrow) * 32];
      bfr.h[0] = *(const v8bf*)&p[ka];
      bfr.h[1] = *(const v8bf*)&p[ka + 16];
      o[nt] = wmma_bf16(pf.v, bfr.v, o[nt]);
    }

    // finish staging next tile (V transpose scatter), then sync
    if (more) {
#pragma unroll
      for (int e = 0; e < 16; ++e) Vt[nb][(kseg * 16 + e) * 32 + krow] = tn.b[e];
    }
    wait_async0();
    __syncthreads();
    cb = nb;
  }

  // normalize + store bf16
#pragma unroll
  for (int nt = 0; nt < 8; ++nt)
#pragma unroll
    for (int r = 0; r < 8; ++r) {
      int row = q0 + r + rhalf;
      int col = nt * 16 + lrow;
      attn[(size_t)row * (NH * D) + (size_t)h * D + col] = (bf16)(o[nt][r] / l[r]);
    }
}

// ---------------------------------------------------------------------------
// Host launcher
// ---------------------------------------------------------------------------
extern "C" void kernel_launch(void* const* d_in, const int* in_sizes, int n_in,
                              void* d_out, int out_size, void* d_ws, size_t ws_size,
                              hipStream_t stream) {
  (void)in_sizes; (void)n_in; (void)out_size; (void)ws_size;
  const int*   positions = (const int*)d_in[0];
  const float* hidden    = (const float*)d_in[1];
  const float* w_qkv     = (const float*)d_in[2];
  const float* w_o       = (const float*)d_in[3];
  const float* qnw       = (const float*)d_in[4];
  const float* knw       = (const float*)d_in[5];
  float* out = (float*)d_out;

  char* ws = (char*)d_ws;
  size_t off = 0;
  bf16* hs_bf   = (bf16*)(ws + off); off += (size_t)T_SEQ * HIDDEN * sizeof(bf16);
  bf16* wqkv_bf = (bf16*)(ws + off); off += (size_t)HIDDEN * QKVN * sizeof(bf16);
  bf16* wo_bf   = (bf16*)(ws + off); off += (size_t)NH * D * HIDDEN * sizeof(bf16);
  float* qkv_f  = (float*)(ws + off); off += (size_t)T_SEQ * QKVN * sizeof(float);
  bf16* q_bf    = (bf16*)(ws + off); off += (size_t)T_SEQ * NH * D * sizeof(bf16);
  bf16* k_bf    = (bf16*)(ws + off); off += (size_t)T_SEQ * NKV * D * sizeof(bf16);
  bf16* v_bf    = (bf16*)(ws + off); off += (size_t)T_SEQ * NKV * D * sizeof(bf16);
  bf16* at_bf   = (bf16*)(ws + off); off += (size_t)T_SEQ * NH * D * sizeof(bf16);

  // 1. downconvert inputs to bf16
  {
    int n4 = (T_SEQ * HIDDEN) / 4;
    cvt_f32_bf16<<<(n4 + 255) / 256, 256, 0, stream>>>(hidden, hs_bf, n4);
  }
  {
    int n4 = (HIDDEN * QKVN) / 4;
    cvt_f32_bf16<<<(n4 + 255) / 256, 256, 0, stream>>>(w_qkv, wqkv_bf, n4);
  }
  {
    int n4 = (NH * D * HIDDEN) / 4;
    cvt_f32_bf16<<<(n4 + 255) / 256, 256, 0, stream>>>(w_o, wo_bf, n4);
  }

  // 2. QKV projection: [T,2048] x [2048,5120] -> f32
  gemm_bf16<<<dim3(QKVN / 128, T_SEQ / 128), 256, 0, stream>>>(
      hs_bf, wqkv_bf, qkv_f, T_SEQ, QKVN, HIDDEN);

  // 3. RMS norm + RoPE -> bf16 Q/K/V
  norm_rope<<<dim3(T_SEQ, 5), 256, 0, stream>>>(positions, qkv_f, qnw, knw,
                                                q_bf, k_bf, v_bf);

  // 4. sliding-window flash attention
  attn_kernel<<<dim3(T_SEQ / 128, NH), 256, 0, stream>>>(q_bf, k_bf, v_bf, at_bf);

  // 5. output projection: [T,4096] x [4096,2048] -> d_out (f32)
  gemm_bf16<<<dim3(HIDDEN / 128, T_SEQ / 128), 256, 0, stream>>>(
      at_bf, wo_bf, out, T_SEQ, HIDDEN, NH * D);
}
